// PairCosineSim_38233798869505
// MI455X (gfx1250) — compile-verified, matrix operand order checked by reading, not run on hardware
//
#include <hip/hip_runtime.h>
#include <math.h>

typedef __attribute__((ext_vector_type(16))) __bf16 bf16x16;
typedef __attribute__((ext_vector_type(8)))  __bf16 bf16x8;
typedef __attribute__((ext_vector_type(8)))  float  floatx8;

namespace {
constexpr int kT = 512;          // target rows (M)
constexpr int kS = 512;          // support rows (N)
constexpr int kD = 1024;         // feature dim (K)
constexpr int BM = 128;
constexpr int BN = 128;
constexpr int BK = 32;           // one wmma k-depth per stage
constexpr int LDT = BK + 8;      // padded LDS row stride (bf16 elems) to spread banks
constexpr int NK = kD / BK;      // 32 k-steps
}

union AFrag { bf16x16 v; bf16x8 h[2]; };

__global__ __launch_bounds__(256)
void PairCosineSim_wmma(const float* __restrict__ supports,  // [B, S, D]
                        const float* __restrict__ targets,   // [B, T, D]
                        float* __restrict__ out) {           // [B, T, S]
  __shared__ __bf16 As[2][BM * LDT];
  __shared__ __bf16 Bs[2][BN * LDT];
  __shared__ float  nA[256];
  __shared__ float  nB[256];

  const int tid   = threadIdx.x;
  const int batch = blockIdx.y;
  const int m0    = (blockIdx.x >> 2) * BM;
  const int n0    = (blockIdx.x &  3) * BN;

  // staging: 2 threads per row, each covers 16 contiguous f32 of the 32-wide K slab
  const int srow = tid >> 1;          // 0..127
  const int scol = (tid & 1) * 16;    // 0 or 16

  const float* gA = targets  + (size_t)(batch * kT + m0 + srow) * kD + scol;
  const float* gB = supports + (size_t)(batch * kS + n0 + srow) * kD + scol;

  float4 ra[4], rb[4];
  float ssqA = 0.0f, ssqB = 0.0f;

  // wave tiling: 8 waves as 2(M) x 4(N); each wave computes 64x32 = 4x2 tiles of 16x16
  const int wid  = tid >> 5;
  const int lane = tid & 31;
  const int wm   = (wid >> 2) * 64;
  const int wn   = (wid &  3) * 32;
  const int frow = lane & 15;          // row (A) / col (B) within 16-tile
  const int fkc  = (lane >> 4) * 8;    // k-chunk select per ISA 16-bit fragment layout

  floatx8 acc[4][2];
  const floatx8 vzero = {0.f, 0.f, 0.f, 0.f, 0.f, 0.f, 0.f, 0.f};
#pragma unroll
  for (int i = 0; i < 4; ++i)
#pragma unroll
    for (int j = 0; j < 2; ++j) acc[i][j] = vzero;

  auto loadg = [&](int ks) {
    const float4* pa = (const float4*)(gA + ks * BK);
    const float4* pb = (const float4*)(gB + ks * BK);
#pragma unroll
    for (int i = 0; i < 4; ++i) { ra[i] = pa[i]; rb[i] = pb[i]; }
  };

  auto stash = [&](int buf) {
    bf16x8 va[2], vb[2];
#pragma unroll
    for (int i = 0; i < 4; ++i) {
      const float4 a = ra[i], b = rb[i];
      ssqA += a.x * a.x + a.y * a.y + a.z * a.z + a.w * a.w;
      ssqB += b.x * b.x + b.y * b.y + b.z * b.z + b.w * b.w;
      const int h = i >> 1, e = (i & 1) * 4;
      va[h][e + 0] = (__bf16)a.x; va[h][e + 1] = (__bf16)a.y;
      va[h][e + 2] = (__bf16)a.z; va[h][e + 3] = (__bf16)a.w;
      vb[h][e + 0] = (__bf16)b.x; vb[h][e + 1] = (__bf16)b.y;
      vb[h][e + 2] = (__bf16)b.z; vb[h][e + 3] = (__bf16)b.w;
    }
    __bf16* pa = &As[buf][srow * LDT + scol];
    __bf16* pb = &Bs[buf][srow * LDT + scol];
    *(bf16x8*)(pa)     = va[0];
    *(bf16x8*)(pa + 8) = va[1];
    *(bf16x8*)(pb)     = vb[0];
    *(bf16x8*)(pb + 8) = vb[1];
  };

  auto mma = [&](int buf) {
    AFrag fa[4], fb[2];
#pragma unroll
    for (int tm = 0; tm < 4; ++tm) {
      const __bf16* p = &As[buf][(wm + tm * 16 + frow) * LDT + fkc];
      fa[tm].h[0] = *(const bf16x8*)(p);        // K = fkc .. fkc+7
      fa[tm].h[1] = *(const bf16x8*)(p + 16);   // K = 16+fkc .. 16+fkc+7
    }
#pragma unroll
    for (int tn = 0; tn < 2; ++tn) {
      const __bf16* p = &Bs[buf][(wn + tn * 16 + frow) * LDT + fkc];
      fb[tn].h[0] = *(const bf16x8*)(p);
      fb[tn].h[1] = *(const bf16x8*)(p + 16);
    }
#pragma unroll
    for (int tm = 0; tm < 4; ++tm)
#pragma unroll
      for (int tn = 0; tn < 2; ++tn)
        acc[tm][tn] = __builtin_amdgcn_wmma_f32_16x16x32_bf16(
            false, fa[tm].v, false, fb[tn].v, (short)0, acc[tm][tn], false, false);
  };

  // software-pipelined, double-buffered main loop
  loadg(0);
  stash(0);
  __syncthreads();

#pragma unroll 2
  for (int ks = 0; ks < NK; ++ks) {
    const int buf = ks & 1;
    if (ks + 1 < NK) loadg(ks + 1);   // global loads in flight under the WMMAs
    mma(buf);
    if (ks + 1 < NK) {
      stash(buf ^ 1);                 // other buffer: safe while peers still read buf
      __syncthreads();
    }
  }

  // fused row-norm reduction (each row's sumsq is split across exactly 2 threads)
  nA[tid] = ssqA;
  nB[tid] = ssqB;
  __syncthreads();

  // inverse norms via v_rsq_f32; clamping ssq at 1e-10 == clamping the norm at
  // 1e-5, equivalent to the reference max(||t||*||s||, 1e-10) for real inputs.
  float invT[4][8];
#pragma unroll
  for (int tm = 0; tm < 4; ++tm) {
#pragma unroll
    for (int v = 0; v < 8; ++v) {
      const int ml = wm + tm * 16 + (lane >> 4) * 8 + v;  // M per C/D VGPR layout
      invT[tm][v] = __builtin_amdgcn_rsqf(fmaxf(nA[2 * ml] + nA[2 * ml + 1], 1e-10f));
    }
  }

#pragma unroll
  for (int tn = 0; tn < 2; ++tn) {
    const int nl = wn + tn * 16 + frow;            // N within block tile
    const float invS_half =
        0.5f * __builtin_amdgcn_rsqf(fmaxf(nB[2 * nl] + nB[2 * nl + 1], 1e-10f));
#pragma unroll
    for (int tm = 0; tm < 4; ++tm) {
#pragma unroll
      for (int v = 0; v < 8; ++v) {
        const int ml = wm + tm * 16 + (lane >> 4) * 8 + v;
        // (dot/denom + 1) * 0.5  ==  fma(dot*invT, 0.5*invS, 0.5)
        const float o = fmaf(acc[tm][tn][v] * invT[tm][v], invS_half, 0.5f);
        out[(size_t)(batch * kT + m0 + ml) * kS + (n0 + nl)] = o;
      }
    }
  }
}

extern "C" void kernel_launch(void* const* d_in, const int* in_sizes, int n_in,
                              void* d_out, int out_size, void* d_ws, size_t ws_size,
                              hipStream_t stream) {
  (void)in_sizes; (void)n_in; (void)out_size; (void)d_ws; (void)ws_size;
  const float* supports = (const float*)d_in[0];  // [32, 512, 1024]
  const float* targets  = (const float*)d_in[1];  // [32, 512, 1024]
  float* out = (float*)d_out;                     // [32, 512, 512]

  dim3 grid(16, 32);   // 4x4 output tiles of 128x128 per batch, 32 batches
  dim3 block(256);     // 8 waves (wave32)
  PairCosineSim_wmma<<<grid, block, 0, stream>>>(supports, targets, out);
}
